// CausalSelfAttention_35759897707310
// MI455X (gfx1250) — matrix operand order repaired and checked by reading, not compiled, not run on hardware
//
#include <hip/hip_runtime.h>

// ---------------------------------------------------------------------------
// Causal self-attention (B=2, T=2048, C=2048, H=16, D=128), bf16 WMMA pipeline
// for gfx1250 (CDNA5, wave32). GEMMs use 2x2 register blocking per wave.
// ---------------------------------------------------------------------------

typedef __attribute__((ext_vector_type(8)))  __bf16 v8bf;
typedef __attribute__((ext_vector_type(16))) __bf16 v16bf;
typedef __attribute__((ext_vector_type(8)))  float  v8f;
typedef unsigned short u16;

static constexpr int BB = 2;
static constexpr int TT = 2048;
static constexpr int CC = 2048;
static constexpr int HH = 16;
static constexpr int DD = 128;
static constexpr float EPS = 1.1920929e-07f;

__device__ __forceinline__ u16 f2bf(float f) {
    unsigned int u = __float_as_uint(f);
    u += 0x7fffu + ((u >> 16) & 1u);          // round-to-nearest-even
    return (u16)(u >> 16);
}

// A-operand (16x32, MxK): lane holds row M=lane%16;
// vector elems e = K = (lane/16)*16 + e  (contiguous, matches load_b)
__device__ __forceinline__ v16bf load_a(const u16* p) {
    v8bf lo = *(const v8bf*)p;
    v8bf hi = *(const v8bf*)(p + 8);
    return __builtin_shufflevector(lo, hi, 0,1,2,3,4,5,6,7,8,9,10,11,12,13,14,15);
}

// B-operand (32x16, KxN): lane holds column N=lane%16;
// vector elems e = K = (lane/16)*16 + e  (one contiguous 32B run)
__device__ __forceinline__ v16bf load_b(const u16* p) {
    return *(const v16bf*)p;
}

__device__ __forceinline__ v8f wmma_bf16(v16bf a, v16bf b, v8f c) {
    return __builtin_amdgcn_wmma_f32_16x16x32_bf16(false, a, false, b,
                                                   (short)0, c, false, false);
}

__device__ __forceinline__ float red16_max(float v) {
#pragma unroll
    for (int m = 1; m < 16; m <<= 1) v = fmaxf(v, __shfl_xor(v, m, 32));
    return v;
}
__device__ __forceinline__ float red16_sum(float v) {
#pragma unroll
    for (int m = 1; m < 16; m <<= 1) v += __shfl_xor(v, m, 32);
    return v;
}
__device__ __forceinline__ float red4_sum(float v) {
#pragma unroll
    for (int m = 1; m < 4; m <<= 1) v += __shfl_xor(v, m, 32);
    return v;
}

// ---------------------------------------------------------------------------
// fp32 -> bf16 conversion
// ---------------------------------------------------------------------------
__global__ void cvt_kernel(const float* __restrict__ src, u16* __restrict__ dst, int n) {
    int i = blockIdx.x * blockDim.x + threadIdx.x;
    int s = gridDim.x * blockDim.x;
    for (; i < n; i += s) dst[i] = f2bf(src[i]);
}

// ---------------------------------------------------------------------------
// Q/K/V projection GEMM (NT, bf16 WMMA, 2x2 blocked) + fused RoPE + RMSNorm.
// grid: (B*T/32, H, 3[q,k,v]); block: 128 threads = 4 waves. Each wave owns a
// 32x32 output block; workgroup covers a 32-token x 128-col (one head) tile.
// ---------------------------------------------------------------------------
__global__ __launch_bounds__(128) void qkv_kernel(
        const u16* __restrict__ xb,
        const u16* __restrict__ wqb, const u16* __restrict__ wkb,
        const u16* __restrict__ wvb,
        const float* __restrict__ cosp, const float* __restrict__ sinp,
        u16* __restrict__ qb, u16* __restrict__ kb, u16* __restrict__ vtb) {
    __shared__ float tile[32][132];

    const int tid  = threadIdx.x;
    const int lane = tid & 31;
    const int w    = tid >> 5;                 // wave 0..3
    const int r    = lane & 15;
    const int hf   = lane >> 4;
    const int m0   = blockIdx.x * 32;
    const int h    = blockIdx.y;
    const int op   = blockIdx.z;               // 0=q 1=k 2=v
    const u16* wb  = (op == 0) ? wqb : (op == 1) ? wkb : wvb;

    const u16* ap0 = xb + (size_t)(m0 + r) * CC;
    const u16* ap1 = xb + (size_t)(m0 + 16 + r) * CC;
    const u16* bp0 = wb + (size_t)(h * 128 + w * 32 + r) * CC;
    const u16* bp1 = wb + (size_t)(h * 128 + w * 32 + 16 + r) * CC;

    v8f acc00, acc01, acc10, acc11;
#pragma unroll
    for (int j = 0; j < 8; ++j) { acc00[j] = 0.f; acc01[j] = 0.f;
                                  acc10[j] = 0.f; acc11[j] = 0.f; }

    for (int k = 0; k < CC; k += 32) {
        v16bf a0 = load_a(ap0 + k + hf * 16);
        v16bf a1 = load_a(ap1 + k + hf * 16);
        v16bf b0 = load_b(bp0 + k + hf * 16);
        v16bf b1 = load_b(bp1 + k + hf * 16);
        acc00 = wmma_bf16(a0, b0, acc00);
        acc01 = wmma_bf16(a0, b1, acc01);
        acc10 = wmma_bf16(a1, b0, acc10);
        acc11 = wmma_bf16(a1, b1, acc11);
    }

    // C layout -> LDS: acc(mi,ni)[j] holds row mi*16 + j + hf*8, col ni*16 + r
#pragma unroll
    for (int j = 0; j < 8; ++j) {
        tile[j + hf * 8][w * 32 + r]            = acc00[j];
        tile[j + hf * 8][w * 32 + 16 + r]       = acc01[j];
        tile[16 + j + hf * 8][w * 32 + r]       = acc10[j];
        tile[16 + j + hf * 8][w * 32 + 16 + r]  = acc11[j];
    }
    __syncthreads();

    // Post-process: thread t owns row t/4, cols (t%4)*32 .. +31
    const int prow = tid >> 2;
    const int cq   = tid & 3;
    const int g    = m0 + prow;
    const int bi   = g / TT;
    const int t    = g % TT;

    float vals[32];
    float sumsq = 0.f;
#pragma unroll
    for (int i = 0; i < 32; ++i) {
        const int c = cq * 32 + i;
        float val;
        if (op == 2) {
            val = tile[prow][c];
        } else {
            if (c < 64) {
                const float cs = cosp[t * 64 + c], sn = sinp[t * 64 + c];
                val =  tile[prow][c] * cs + tile[prow][c + 64] * sn;
            } else {
                const int d2 = c - 64;
                const float cs = cosp[t * 64 + d2], sn = sinp[t * 64 + d2];
                val = -tile[prow][d2] * sn + tile[prow][c] * cs;
            }
            sumsq += val * val;
        }
        vals[i] = val;
    }

    if (op != 2) {
        sumsq = red4_sum(sumsq);                     // full row (D=128)
        const float rn = rsqrtf(sumsq * (1.f / 128.f) + EPS);
        u16* outp = ((op == 0) ? qb : kb) +
                    (((size_t)(bi * HH + h) * TT + t) * DD + cq * 32);
#pragma unroll
        for (int i = 0; i < 32; ++i) outp[i] = f2bf(vals[i] * rn);
    } else {
        // V transposed: (B,H,D,T)
        const size_t basev = ((size_t)(bi * HH + h) * DD) * TT;
#pragma unroll
        for (int i = 0; i < 32; ++i)
            vtb[basev + (size_t)(cq * 32 + i) * TT + t] = f2bf(vals[i]);
    }
}

// ---------------------------------------------------------------------------
// Flash attention: one wave per (b, h, 16-row q-tile). Streams 32 keys/iter.
// ---------------------------------------------------------------------------
__global__ __launch_bounds__(32) void attn_kernel(
        const u16* __restrict__ qb, const u16* __restrict__ kb,
        const u16* __restrict__ vtb, u16* __restrict__ yb) {
    __shared__ __align__(32) u16 plds[16 * 32];

    const int lane = threadIdx.x;
    const int r    = lane & 15;
    const int hf   = lane >> 4;
    const int q0   = blockIdx.x * 16;
    const int h    = blockIdx.y;
    const int bbk  = blockIdx.z;

    const u16* qp = qb  + ((size_t)(bbk * HH + h) * TT) * DD;
    const u16* kp = kb  + ((size_t)(bbk * HH + h) * TT) * DD;
    const u16* vp = vtb + ((size_t)(bbk * HH + h) * DD) * TT;

    // Hold full 16x128 Q tile in A-operand registers (4 K-steps of 32)
    v16bf aQ[4];
#pragma unroll
    for (int kd = 0; kd < 4; ++kd)
        aQ[kd] = load_a(qp + (size_t)(q0 + r) * DD + kd * 32 + hf * 16);

    v8f O[8];
    float rmax[8], rsum[8], fsc[8];
#pragma unroll
    for (int dt = 0; dt < 8; ++dt)
#pragma unroll
        for (int j = 0; j < 8; ++j) O[dt][j] = 0.f;
#pragma unroll
    for (int j = 0; j < 8; ++j) { rmax[j] = -1e30f; rsum[j] = 0.f; }

    const float scale = 0.08838834764831845f;   // 1/sqrt(128)
    const int kend = q0 + 16;                    // exclusive causal limit

    for (int kt = 0; kt < kend; kt += 32) {
        v8f s0, s1;
#pragma unroll
        for (int j = 0; j < 8; ++j) { s0[j] = 0.f; s1[j] = 0.f; }

        // next-tile prefetch (global_prefetch_b8) to hide L2 latency
        __builtin_prefetch(kp + (size_t)(kt + 32 + r) * DD, 0, 3);
        __builtin_prefetch(vp + (size_t)r * TT + kt + 32, 0, 3);

#pragma unroll
        for (int kd = 0; kd < 4; ++kd) {
            v16bf b0 = load_b(kp + (size_t)(kt + r)      * DD + kd * 32 + hf * 16);
            v16bf b1 = load_b(kp + (size_t)(kt + 16 + r) * DD + kd * 32 + hf * 16);
            s0 = wmma_bf16(aQ[kd], b0, s0);
            s1 = wmma_bf16(aQ[kd], b1, s1);
        }

        // Online softmax, rows m = j + hf*8 live in one 16-lane half
#pragma unroll
        for (int j = 0; j < 8; ++j) {
            const int m    = j + hf * 8;
            const int qrow = q0 + m;
            const int c0   = kt + r;
            float e0 = s0[j] * scale;
            float e1 = s1[j] * scale;
            if (c0 > qrow)      e0 = -1e30f;
            if (c0 + 16 > qrow) e1 = -1e30f;
            const float mx = red16_max(fmaxf(e0, e1));
            const float nm = fmaxf(rmax[j], mx);
            const float f  = __expf(rmax[j] - nm);
            const float p0 = __expf(e0 - nm);
            const float p1 = __expf(e1 - nm);
            rsum[j] = rsum[j] * f + red16_sum(p0 + p1);
            rmax[j] = nm;
            fsc[j]  = f;
            plds[m * 32 + r]      = f2bf(p0);
            plds[m * 32 + 16 + r] = f2bf(p1);
        }
#pragma unroll
        for (int dt = 0; dt < 8; ++dt)
#pragma unroll
            for (int j = 0; j < 8; ++j) O[dt][j] *= fsc[j];

        __syncthreads();
        v16bf pA = load_a(plds + r * 32 + hf * 16);   // P tile as A operand
#pragma unroll
        for (int dt = 0; dt < 8; ++dt) {
            v16bf vB = load_b(vp + (size_t)(dt * 16 + r) * TT + kt + hf * 16);
            O[dt] = wmma_bf16(pA, vB, O[dt]);
        }
        __syncthreads();
    }

    // Normalize and write attention output (B,T,C) bf16
#pragma unroll
    for (int j = 0; j < 8; ++j) {
        const float inv = 1.f / rsum[j];
        const int g = q0 + j + hf * 8;
        const size_t base = ((size_t)(bbk * TT + g)) * CC + h * DD;
#pragma unroll
        for (int dt = 0; dt < 8; ++dt)
            yb[base + dt * 16 + r] = f2bf(O[dt][j] * inv);
    }
}

// ---------------------------------------------------------------------------
// Output projection: y @ wproj^T  (NT GEMM, 2x2 blocked, fp32 out)
// grid: (B*T/32, C/128); block: 128 = 4 waves, each wave one 32x32 block
// ---------------------------------------------------------------------------
__global__ __launch_bounds__(128) void proj_kernel(
        const u16* __restrict__ yb, const u16* __restrict__ wpb,
        float* __restrict__ out) {
    const int tid  = threadIdx.x;
    const int lane = tid & 31;
    const int w    = tid >> 5;
    const int r    = lane & 15;
    const int hf   = lane >> 4;
    const int m0   = blockIdx.x * 32;
    const int n0   = blockIdx.y * 128;

    const u16* ap0 = yb  + (size_t)(m0 + r) * CC;
    const u16* ap1 = yb  + (size_t)(m0 + 16 + r) * CC;
    const u16* bp0 = wpb + (size_t)(n0 + w * 32 + r) * CC;
    const u16* bp1 = wpb + (size_t)(n0 + w * 32 + 16 + r) * CC;

    v8f acc00, acc01, acc10, acc11;
#pragma unroll
    for (int j = 0; j < 8; ++j) { acc00[j] = 0.f; acc01[j] = 0.f;
                                  acc10[j] = 0.f; acc11[j] = 0.f; }

    for (int k = 0; k < CC; k += 32) {
        v16bf a0 = load_a(ap0 + k + hf * 16);
        v16bf a1 = load_a(ap1 + k + hf * 16);
        v16bf b0 = load_b(bp0 + k + hf * 16);
        v16bf b1 = load_b(bp1 + k + hf * 16);
        acc00 = wmma_bf16(a0, b0, acc00);
        acc01 = wmma_bf16(a0, b1, acc01);
        acc10 = wmma_bf16(a1, b0, acc10);
        acc11 = wmma_bf16(a1, b1, acc11);
    }

#pragma unroll
    for (int j = 0; j < 8; ++j) {
        const size_t row0 = (size_t)(m0 + j + hf * 8) * CC;
        const size_t row1 = (size_t)(m0 + 16 + j + hf * 8) * CC;
        out[row0 + n0 + w * 32 + r]      = acc00[j];
        out[row0 + n0 + w * 32 + 16 + r] = acc01[j];
        out[row1 + n0 + w * 32 + r]      = acc10[j];
        out[row1 + n0 + w * 32 + 16 + r] = acc11[j];
    }
}

// ---------------------------------------------------------------------------
extern "C" void kernel_launch(void* const* d_in, const int* in_sizes, int n_in,
                              void* d_out, int out_size, void* d_ws, size_t ws_size,
                              hipStream_t stream) {
    const float* x    = (const float*)d_in[0];
    const float* cosp = (const float*)d_in[1];
    const float* sinp = (const float*)d_in[2];
    const float* wq   = (const float*)d_in[3];
    const float* wk   = (const float*)d_in[4];
    const float* wv   = (const float*)d_in[5];
    const float* wp   = (const float*)d_in[6];
    float* out = (float*)d_out;

    const size_t NX = (size_t)BB * TT * CC;   // 8Mi
    const size_t NW = (size_t)CC * CC;        // 4Mi

    u16* xb  = (u16*)d_ws;
    u16* wqb = xb  + NX;
    u16* wkb = wqb + NW;
    u16* wvb = wkb + NW;
    u16* wpb = wvb + NW;
    u16* qb  = wpb + NW;
    u16* kb  = qb  + NX;
    u16* vtb = kb  + NX;
    u16* yb  = vtb + NX;
    (void)ws_size; (void)in_sizes; (void)n_in; (void)out_size;

    cvt_kernel<<<4096, 256, 0, stream>>>(x,  xb,  (int)NX);
    cvt_kernel<<<2048, 256, 0, stream>>>(wq, wqb, (int)NW);
    cvt_kernel<<<2048, 256, 0, stream>>>(wk, wkb, (int)NW);
    cvt_kernel<<<2048, 256, 0, stream>>>(wv, wvb, (int)NW);
    cvt_kernel<<<2048, 256, 0, stream>>>(wp, wpb, (int)NW);

    qkv_kernel<<<dim3(BB * TT / 32, HH, 3), 128, 0, stream>>>(
        xb, wqb, wkb, wvb, cosp, sinp, qb, kb, vtb);

    attn_kernel<<<dim3(TT / 16, HH, BB), 32, 0, stream>>>(qb, kb, vtb, yb);

    proj_kernel<<<dim3(BB * TT / 32, CC / 128), 128, 0, stream>>>(yb, wpb, out);
}
